// RelationModel0_67628555043036
// MI455X (gfx1250) — matrix operand order repaired and checked by reading, not compiled, not run on hardware
//
#include <hip/hip_runtime.h>

// ---------------- problem constants ----------------
#define Bsz   256
#define Nn    100
#define OBJD  2048
#define Ll    14
#define EMBD  300
#define EMBP  320      // K pad for GRU input GEMM (mult of 32)
#define Hh    1024
#define NANSN 3129
#define NANSP 3200     // pad to multiple of 128

typedef __attribute__((ext_vector_type(16))) __bf16 v16bf;
typedef __attribute__((ext_vector_type(8)))  __bf16 v8bf;
typedef __attribute__((ext_vector_type(8)))  float  v8f;
typedef __attribute__((ext_vector_type(4)))  int    v4i;

// ---------------- async global->LDS copy (CDNA5) ----------------
#if defined(__AMDGCN__) && __has_builtin(__builtin_amdgcn_global_load_async_to_lds_b128)
#define HAS_ASYNC_LDS 1
typedef __attribute__((address_space(1))) v4i gv4i;   // printed by clang as "v4i __device__*"
typedef __attribute__((address_space(3))) v4i lv4i;   // printed by clang as "v4i __shared__*"
#endif

__device__ __forceinline__ void copy16_g2l(const __bf16* g, __bf16* l)
{
#ifdef HAS_ASYNC_LDS
    // global_load_async_to_lds_b128: ASYNCcnt-tracked DMA, no VGPR round-trip.
    __builtin_amdgcn_global_load_async_to_lds_b128(
        (gv4i*)(uintptr_t)g,
        (lv4i*)(uint32_t)(uintptr_t)l,   // low 32 bits of flat shared addr = LDS offset
        0, 0);
#else
    *(v8bf*)l = *(const v8bf*)g;
#endif
}

__device__ __forceinline__ void async_wait_all()
{
#ifdef HAS_ASYNC_LDS
    asm volatile("s_wait_asynccnt 0x0" ::: "memory");
#endif
}

// =====================================================================
// Generic batched bf16 WMMA GEMM:  C[b] = act( A[b] (M x K) * BT[b]^T + bias [+ C] )
// A row-major (lda = K), BT row-major (N x K, ldb = K), C row-major (ldc).
// Block: 256 threads (8 waves), tile 64(M) x 128(N); wave tile 32x32.
// Double-buffered LDS staging: async copies of the 64x32 A-tile and 128x32
// B-tile overlap with WMMA on the previous k-step.
// Requires: K % 32 == 0, padded A rows up to gridDim.x*64, padded BT rows
// up to gridDim.y*128.  Stores guarded by Mreal/Nreal.
// =====================================================================
__launch_bounds__(256)
__global__ void gemm_bf16_wmma(const __bf16* __restrict__ A,
                               const __bf16* __restrict__ BT,
                               float* __restrict__ C,
                               const float* __restrict__ bias,
                               int Mreal, int Nreal, int K, int ldc,
                               long long sA, long long sB, long long sC,
                               int act, int beta)
{
    __shared__ __attribute__((aligned(16))) __bf16 smA[2][64 * 32];   //  4 KB x2
    __shared__ __attribute__((aligned(16))) __bf16 smB[2][128 * 32];  //  8 KB x2

    const long long bz = blockIdx.z;
    A  += bz * sA;
    BT += bz * sB;
    C  += bz * sC;

    const int tid   = threadIdx.x;
    const int lane  = tid & 31;
    const int wave  = tid >> 5;
    const int wm    = wave & 1;          // 2 waves along M
    const int wn    = wave >> 1;         // 4 waves along N
    const int m0b   = blockIdx.x * 64;
    const int n0b   = blockIdx.y * 128;
    const int lmod  = lane & 15;
    const int lhalf = lane >> 4;

    // cooperative copy mapping (16B chunks)
    const int arow = tid >> 2;           // 0..63
    const int acol = (tid & 3) * 8;      // 0,8,16,24
    const int brow0 = tid >> 1;          // 0..127  (chunk set 0: cols 0..15)
    const int bcol0 = (tid & 1) * 8;
    const int brow1 = brow0;             // chunk set 1: cols 16..31
    const int bcol1 = bcol0 + 16;

    const __bf16* gA  = A  + (size_t)(m0b + arow) * K + acol;
    const __bf16* gB0 = BT + (size_t)(n0b + brow0) * K + bcol0;
    const __bf16* gB1 = BT + (size_t)(n0b + brow1) * K + bcol1;

    const int nsteps = K >> 5;

    // prologue: stage k-step 0 into buffer 0
    copy16_g2l(gA,  &smA[0][arow * 32 + acol]);
    copy16_g2l(gB0, &smB[0][brow0 * 32 + bcol0]);
    copy16_g2l(gB1, &smB[0][brow1 * 32 + bcol1]);

    v8f acc00 = {}, acc01 = {}, acc10 = {}, acc11 = {};

    for (int kt = 0; kt < nsteps; ++kt) {
        const int cur = kt & 1;
        async_wait_all();
        __syncthreads();

        if (kt + 1 < nsteps) {
            const int nxt = cur ^ 1;
            const int ko = (kt + 1) * 32;
            copy16_g2l(gA  + ko, &smA[nxt][arow * 32 + acol]);
            copy16_g2l(gB0 + ko, &smB[nxt][brow0 * 32 + bcol0]);
            copy16_g2l(gB1 + ko, &smB[nxt][brow1 * 32 + bcol1]);
        }

        // A fragments: lane half selects contiguous 16-K slice of row m.
        const __bf16* la0 = &smA[cur][(wm * 32 + lmod) * 32];
        const __bf16* la1 = la0 + 16 * 32;
        v8bf a0lo = *(const v8bf*)(la0 + lhalf * 16);
        v8bf a0hi = *(const v8bf*)(la0 + 8 + lhalf * 16);
        v8bf a1lo = *(const v8bf*)(la1 + lhalf * 16);
        v8bf a1hi = *(const v8bf*)(la1 + 8 + lhalf * 16);
        // B fragments: lane half selects contiguous 16-K slice of column n.
        v16bf b0 = *(const v16bf*)(&smB[cur][(wn * 32 + lmod) * 32 + lhalf * 16]);
        v16bf b1 = *(const v16bf*)(&smB[cur][(wn * 32 + 16 + lmod) * 32 + lhalf * 16]);

        v16bf a0, a1;
#pragma unroll
        for (int i = 0; i < 8; ++i) {
            a0[i] = a0lo[i]; a0[i + 8] = a0hi[i];
            a1[i] = a1lo[i]; a1[i + 8] = a1hi[i];
        }

        acc00 = __builtin_amdgcn_wmma_f32_16x16x32_bf16(false, a0, false, b0, (short)0, acc00, false, false);
        acc01 = __builtin_amdgcn_wmma_f32_16x16x32_bf16(false, a0, false, b1, (short)0, acc01, false, false);
        acc10 = __builtin_amdgcn_wmma_f32_16x16x32_bf16(false, a1, false, b0, (short)0, acc10, false, false);
        acc11 = __builtin_amdgcn_wmma_f32_16x16x32_bf16(false, a1, false, b1, (short)0, acc11, false, false);
    }

    const int m0 = m0b + wm * 32;
    const int n0 = n0b + wn * 32;
    v8f accs[2][2] = {{acc00, acc01}, {acc10, acc11}};
#pragma unroll
    for (int fm = 0; fm < 2; ++fm) {
#pragma unroll
        for (int fn = 0; fn < 2; ++fn) {
            const int nIdx  = n0 + fn * 16 + lmod;
            const int mBase = m0 + fm * 16 + lhalf * 8;   // C layout: VGPR r -> M = base + r
#pragma unroll
            for (int r = 0; r < 8; ++r) {
                const int mIdx = mBase + r;
                if (mIdx < Mreal && nIdx < Nreal) {
                    float vv = accs[fm][fn][r];
                    if (bias) vv += bias[nIdx];
                    if (beta) vv += C[(size_t)mIdx * ldc + nIdx];
                    if (act == 1)      vv = fmaxf(vv, 0.f);
                    else if (act == 2) vv = tanhf(vv);
                    C[(size_t)mIdx * ldc + nIdx] = vv;
                }
            }
        }
    }
}

// ---------------- helper / elementwise kernels ----------------

// Transpose W (R x C) -> WT (Cpad x Rpad) bf16, zero padded.  total = Cpad*Rpad.
__global__ void k_wT(const float* __restrict__ W, __bf16* __restrict__ WT,
                     int R, int C, int Rpad, long long total)
{
    long long i = (long long)blockIdx.x * 256 + threadIdx.x;
    if (i >= total) return;
    int r = (int)(i % Rpad);
    long long c = i / Rpad;
    WT[i] = (r < R && c < C) ? (__bf16)W[(size_t)r * C + c] : (__bf16)0.f;
}

__global__ void k_cvt(const float* __restrict__ s, __bf16* __restrict__ d, long long n)
{
    long long i = (long long)blockIdx.x * 256 + threadIdx.x;
    if (i < n) d[i] = (__bf16)s[i];
}

__global__ void k_add_cvt(const float* __restrict__ a, const float* __restrict__ b,
                          __bf16* __restrict__ d, long long n)
{
    long long i = (long long)blockIdx.x * 256 + threadIdx.x;
    if (i < n) d[i] = (__bf16)(a[i] + b[i]);
}

__global__ void k_mul_cvt(const float* __restrict__ a, const float* __restrict__ b,
                          __bf16* __restrict__ d, long long n)
{
    long long i = (long long)blockIdx.x * 256 + threadIdx.x;
    if (i < n) d[i] = (__bf16)(a[i] * b[i]);
}

// Embedding gather with K padding: X (B*L x EMBP) bf16.
__global__ void k_embed(const float* __restrict__ emb, const int* __restrict__ q,
                        __bf16* __restrict__ X)
{
    long long i = (long long)blockIdx.x * 256 + threadIdx.x;
    const long long total = (long long)Bsz * Ll * EMBP;
    if (i >= total) return;
    int k = (int)(i % EMBP);
    long long row = i / EMBP;
    X[i] = (k < EMBD) ? (__bf16)emb[(size_t)q[row] * EMBD + k] : (__bf16)0.f;
}

// GRU gate nonlinearity for step t.
__global__ void k_gru_gate(const float* __restrict__ GI, const float* __restrict__ GH,
                           float* __restrict__ h, __bf16* __restrict__ hbf, int t)
{
    int i = blockIdx.x * 256 + threadIdx.x;
    if (i >= Bsz * Hh) return;
    int b = i / Hh, j = i % Hh;
    const float* gi = GI + ((size_t)b * Ll + t) * 3 * Hh;
    const float* gh = GH + (size_t)b * 3 * Hh;
    float ir = gi[j], iz = gi[Hh + j], in = gi[2 * Hh + j];
    float hr = gh[j], hz = gh[Hh + j], hn = gh[2 * Hh + j];
    float r  = 1.f / (1.f + expf(-(ir + hr)));
    float z  = 1.f / (1.f + expf(-(iz + hz)));
    float nv = tanhf(in + r * hn);
    float hv = (1.f - z) * nv + z * h[i];
    h[i] = hv;
    hbf[i] = (__bf16)hv;
}

// s_proj = tanh(b[:,:, :4] @ ws_b); also PQ = s_proj * s_q.  Padded to 128 rows.
__global__ void k_sproj(const float* __restrict__ bbox, const float* __restrict__ wsb,
                        const float* __restrict__ sq,
                        __bf16* __restrict__ P, __bf16* __restrict__ PQ)
{
    long long i = (long long)blockIdx.x * 256 + threadIdx.x;
    const long long total = (long long)Bsz * 128 * Hh;
    if (i >= total) return;
    int hh = (int)(i % Hh);
    long long t = i / Hh;
    int n = (int)(t % 128), b = (int)(t / 128);
    if (n < Nn) {
        const float* bbp = bbox + ((size_t)b * Nn + n) * 6;
        float a = bbp[0] * wsb[hh] + bbp[1] * wsb[Hh + hh]
                + bbp[2] * wsb[2 * Hh + hh] + bbp[3] * wsb[3 * Hh + hh];
        float p = tanhf(a);
        P[i]  = (__bf16)p;
        PQ[i] = (__bf16)(p * sq[(size_t)b * Hh + hh]);
    } else {
        P[i] = (__bf16)0.f; PQ[i] = (__bf16)0.f;
    }
}

// Pack v_proj (tanh already applied by GEMM) into padded batched layout + *v_q copy.
__global__ void k_vpack(const float* __restrict__ vproj, const float* __restrict__ vq,
                        __bf16* __restrict__ VP, __bf16* __restrict__ VPQ)
{
    long long i = (long long)blockIdx.x * 256 + threadIdx.x;
    const long long total = (long long)Bsz * 128 * Hh;
    if (i >= total) return;
    int hh = (int)(i % Hh);
    long long t = i / Hh;
    int n = (int)(t % 128), b = (int)(t / 128);
    if (n < Nn) {
        float p = vproj[((size_t)b * Nn + n) * Hh + hh];
        VP[i]  = (__bf16)p;
        VPQ[i] = (__bf16)(p * vq[(size_t)b * Hh + hh]);
    } else {
        VP[i] = (__bf16)0.f; VPQ[i] = (__bf16)0.f;
    }
}

// v^T packed: (B, OBJD, 128) bf16, zero pad n>=100.
__global__ void k_vT(const float* __restrict__ v, __bf16* __restrict__ vT)
{
    long long i = (long long)blockIdx.x * 256 + threadIdx.x;
    const long long total = (long long)Bsz * OBJD * 128;
    if (i >= total) return;
    int n = (int)(i % 128);
    long long t = i / 128;
    int d = (int)(t % OBJD), b = (int)(t / OBJD);
    vT[i] = (n < Nn) ? (__bf16)v[((size_t)b * Nn + n) * OBJD + d] : (__bf16)0.f;
}

// Row softmax over last dim (100) of rel logits -> padded 128x128 bf16.
__global__ void k_rel_softmax(const float* __restrict__ logits, __bf16* __restrict__ relbf)
{
    int w    = (blockIdx.x * 256 + threadIdx.x) >> 5;
    int lane = threadIdx.x & 31;
    if (w >= Bsz * 128) return;
    int b = w >> 7, rr = w & 127;
    __bf16* dst = relbf + ((size_t)b * 128 + rr) * 128;
    if (rr < Nn) {
        const float* row = logits + ((size_t)b * Nn + rr) * Nn;
        float x[4]; int cnt = 0; float mx = -1e30f;
        for (int c = lane; c < Nn; c += 32) { x[cnt] = row[c]; mx = fmaxf(mx, x[cnt]); ++cnt; }
        for (int o = 16; o > 0; o >>= 1) mx = fmaxf(mx, __shfl_xor(mx, o));
        float s = 0.f;
        for (int i = 0; i < cnt; ++i) { x[i] = expf(x[i] - mx); s += x[i]; }
        for (int o = 16; o > 0; o >>= 1) s += __shfl_xor(s, o);
        float inv = 1.f / s; cnt = 0;
        for (int c = lane; c < Nn; c += 32) { dst[c] = (__bf16)(x[cnt] * inv); ++cnt; }
        for (int c = Nn + lane; c < 128; c += 32) dst[c] = (__bf16)0.f;
    } else {
        for (int c = lane; c < 128; c += 32) dst[c] = (__bf16)0.f;
    }
}

// att_logits[b,n] = sum_h av[b,n,h] * aq[b,h] * wa_out[h]  (av, aq already ReLU'd)
__global__ void k_att_logits(const float* __restrict__ av, const float* __restrict__ aq,
                             const float* __restrict__ waout, float* __restrict__ logits)
{
    int w    = (blockIdx.x * 256 + threadIdx.x) >> 5;
    int lane = threadIdx.x & 31;
    if (w >= Bsz * Nn) return;
    int b = w / Nn;
    const float* arow = av + (size_t)w * Hh;
    const float* qrow = aq + (size_t)b * Hh;
    float s = 0.f;
    for (int h = lane; h < Hh; h += 32) s += arow[h] * qrow[h] * waout[h];
    for (int o = 16; o > 0; o >>= 1) s += __shfl_xor(s, o);
    if (lane == 0) logits[w] = s;
}

// softmax over n (=100) per batch row, in place.
__global__ void k_att_softmax(float* __restrict__ logits)
{
    int w    = (blockIdx.x * 256 + threadIdx.x) >> 5;
    int lane = threadIdx.x & 31;
    if (w >= Bsz) return;
    float* row = logits + (size_t)w * Nn;
    float x[4]; int cnt = 0; float mx = -1e30f;
    for (int c = lane; c < Nn; c += 32) { x[cnt] = row[c]; mx = fmaxf(mx, x[cnt]); ++cnt; }
    for (int o = 16; o > 0; o >>= 1) mx = fmaxf(mx, __shfl_xor(mx, o));
    float s = 0.f;
    for (int i = 0; i < cnt; ++i) { x[i] = expf(x[i] - mx); s += x[i]; }
    for (int o = 16; o > 0; o >>= 1) s += __shfl_xor(s, o);
    float inv = 1.f / s; cnt = 0;
    for (int c = lane; c < Nn; c += 32) { row[c] = x[cnt] * inv; ++cnt; }
}

// v_emb[b,d] = sum_n att[b,n] * (v[b,n,d] + v_nbr[b,n,d])  -> bf16
__global__ void k_vemb(const float* __restrict__ v, const float* __restrict__ vnbr,
                       const float* __restrict__ att, __bf16* __restrict__ vembbf)
{
    long long i = (long long)blockIdx.x * 256 + threadIdx.x;
    if (i >= (long long)Bsz * OBJD) return;
    int b = (int)(i / OBJD), d = (int)(i % OBJD);
    float s = 0.f;
    for (int n = 0; n < Nn; ++n) {
        size_t idx = ((size_t)b * Nn + n) * OBJD + d;
        s += att[b * Nn + n] * (v[idx] + vnbr[idx]);
    }
    vembbf[i] = (__bf16)s;
}

// =====================================================================
extern "C" void kernel_launch(void* const* d_in, const int* in_sizes, int n_in,
                              void* d_out, int out_size, void* d_ws, size_t ws_size,
                              hipStream_t stream)
{
    (void)in_sizes; (void)n_in; (void)out_size; (void)ws_size;

    const float* v      = (const float*)d_in[0];
    const float* bbox   = (const float*)d_in[1];
    const int*   q      = (const int*)  d_in[3];
    const float* emb    = (const float*)d_in[5];
    const float* wih    = (const float*)d_in[6];
    const float* whh    = (const float*)d_in[7];
    const float* bih    = (const float*)d_in[8];
    const float* bhh    = (const float*)d_in[9];
    const float* wq_net = (const float*)d_in[10];
    const float* bq_net = (const float*)d_in[11];
    const float* ws_b   = (const float*)d_in[12];
    const float* ws_q   = (const float*)d_in[13];
    const float* wv_v   = (const float*)d_in[14];
    const float* wv_q   = (const float*)d_in[15];
    const float* wa_v   = (const float*)d_in[16];
    const float* wa_q   = (const float*)d_in[17];
    const float* wa_out = (const float*)d_in[18];
    const float* wv_net = (const float*)d_in[19];
    const float* bv_net = (const float*)d_in[20];
    const float* wc1    = (const float*)d_in[21];
    const float* bc1    = (const float*)d_in[22];
    const float* wc2    = (const float*)d_in[23];
    const float* bc2    = (const float*)d_in[24];
    float* out = (float*)d_out;

    // ---- workspace bump allocator (256B aligned) ----
    size_t off = 0;
    auto alloc = [&](size_t bytes) -> void* {
        void* p = (char*)d_ws + off;
        off += (bytes + 255) & ~(size_t)255;
        return p;
    };

    __bf16* WIHT  = (__bf16*)alloc((size_t)3 * Hh * EMBP * 2);
    __bf16* WHHT  = (__bf16*)alloc((size_t)3 * Hh * Hh * 2);
    __bf16* WQT   = (__bf16*)alloc((size_t)Hh * Hh * 2);
    __bf16* WSQT  = (__bf16*)alloc((size_t)Hh * Hh * 2);
    __bf16* WVQT  = (__bf16*)alloc((size_t)Hh * Hh * 2);
    __bf16* WAQT  = (__bf16*)alloc((size_t)Hh * Hh * 2);
    __bf16* WVVT  = (__bf16*)alloc((size_t)Hh * OBJD * 2);
    __bf16* WAVT  = (__bf16*)alloc((size_t)Hh * OBJD * 2);
    __bf16* WVNT  = (__bf16*)alloc((size_t)Hh * OBJD * 2);
    __bf16* WC1T  = (__bf16*)alloc((size_t)2 * Hh * Hh * 2);
    __bf16* WC2T  = (__bf16*)alloc((size_t)NANSP * OBJD * 2);
    __bf16* Xbf   = (__bf16*)alloc((size_t)Bsz * Ll * EMBP * 2);
    float*  GI    = (float*) alloc((size_t)Bsz * Ll * 3 * Hh * 4);
    float*  GH    = (float*) alloc((size_t)Bsz * 3 * Hh * 4);
    float*  hbuf  = (float*) alloc((size_t)Bsz * Hh * 4);
    __bf16* hbf   = (__bf16*)alloc((size_t)Bsz * Hh * 2);
    float*  qrep  = (float*) alloc((size_t)Bsz * Hh * 4);
    float*  sq    = (float*) alloc((size_t)Bsz * Hh * 4);
    float*  vq    = (float*) alloc((size_t)Bsz * Hh * 4);
    float*  aqb   = (float*) alloc((size_t)Bsz * Hh * 4);
    __bf16* vbf   = (__bf16*)alloc((size_t)Bsz * Nn * OBJD * 2);  // later reused as v2_bf
    float*  vproj = (float*) alloc((size_t)Bsz * Nn * Hh * 4);    // later reused as av
    __bf16* Pbf   = (__bf16*)alloc((size_t)Bsz * 128 * Hh * 2);
    __bf16* PQbf  = (__bf16*)alloc((size_t)Bsz * 128 * Hh * 2);
    __bf16* VPbf  = (__bf16*)alloc((size_t)Bsz * 128 * Hh * 2);
    __bf16* VPQbf = (__bf16*)alloc((size_t)Bsz * 128 * Hh * 2);
    float*  rlog  = (float*) alloc((size_t)Bsz * Nn * Nn * 4);
    __bf16* relbf = (__bf16*)alloc((size_t)Bsz * 128 * 128 * 2);
    __bf16* vTbf  = (__bf16*)alloc((size_t)Bsz * OBJD * 128 * 2);
    float*  vnbr  = (float*) alloc((size_t)Bsz * Nn * OBJD * 4);
    float*  attl  = (float*) alloc((size_t)Bsz * Nn * 4);
    __bf16* vembbf= (__bf16*)alloc((size_t)Bsz * OBJD * 2);
    float*  vrep  = (float*) alloc((size_t)Bsz * Hh * 4);
    __bf16* jntbf = (__bf16*)alloc((size_t)Bsz * Hh * 2);
    float*  hid   = (float*) alloc((size_t)Bsz * 2 * Hh * 4);
    __bf16* hidbf = (__bf16*)alloc((size_t)Bsz * 2 * Hh * 2);

    dim3 blk(256);
    auto g1 = [](long long n) { return dim3((unsigned)((n + 255) / 256)); };

    auto gemm = [&](const __bf16* A, const __bf16* BT, float* C, const float* bias,
                    int Mpad, int Mreal, int Npad, int Nreal, int K, int ldc,
                    long long sA, long long sB, long long sC, int batch,
                    int act, int beta) {
        dim3 grid(Mpad / 64, Npad / 128, batch);
        gemm_bf16_wmma<<<grid, blk, 0, stream>>>(A, BT, C, bias, Mreal, Nreal, K, ldc,
                                                 sA, sB, sC, act, beta);
    };

    // ---- weight transposes / conversions to bf16 ----
    k_wT<<<g1((long long)3*Hh*EMBP), blk, 0, stream>>>(wih,    WIHT, EMBD, 3*Hh, EMBP, (long long)3*Hh*EMBP);
    k_wT<<<g1((long long)3*Hh*Hh),   blk, 0, stream>>>(whh,    WHHT, Hh,   3*Hh, Hh,   (long long)3*Hh*Hh);
    k_wT<<<g1((long long)Hh*Hh),     blk, 0, stream>>>(wq_net, WQT,  Hh,   Hh,   Hh,   (long long)Hh*Hh);
    k_wT<<<g1((long long)Hh*Hh),     blk, 0, stream>>>(ws_q,   WSQT, Hh,   Hh,   Hh,   (long long)Hh*Hh);
    k_wT<<<g1((long long)Hh*Hh),     blk, 0, stream>>>(wv_q,   WVQT, Hh,   Hh,   Hh,   (long long)Hh*Hh);
    k_wT<<<g1((long long)Hh*Hh),     blk, 0, stream>>>(wa_q,   WAQT, Hh,   Hh,   Hh,   (long long)Hh*Hh);
    k_wT<<<g1((long long)Hh*OBJD),   blk, 0, stream>>>(wv_v,   WVVT, OBJD, Hh,   OBJD, (long long)Hh*OBJD);
    k_wT<<<g1((long long)Hh*OBJD),   blk, 0, stream>>>(wa_v,   WAVT, OBJD, Hh,   OBJD, (long long)Hh*OBJD);
    k_wT<<<g1((long long)Hh*OBJD),   blk, 0, stream>>>(wv_net, WVNT, OBJD, Hh,   OBJD, (long long)Hh*OBJD);
    k_wT<<<g1((long long)2*Hh*Hh),   blk, 0, stream>>>(wc1,    WC1T, Hh,   2*Hh, Hh,   (long long)2*Hh*Hh);
    k_wT<<<g1((long long)NANSP*OBJD),blk, 0, stream>>>(wc2,    WC2T, OBJD, NANSN,OBJD, (long long)NANSP*OBJD);

    // ---- question path: embedding + GRU ----
    k_embed<<<g1((long long)Bsz*Ll*EMBP), blk, 0, stream>>>(emb, q, Xbf);
    (void)hipMemsetAsync(hbuf, 0, (size_t)Bsz * Hh * 4, stream);
    (void)hipMemsetAsync(hbf,  0, (size_t)Bsz * Hh * 2, stream);

    // GI = X @ wih + bih for all timesteps at once: (B*L x 320) * (320 x 3072)
    gemm(Xbf, WIHT, GI, bih, Bsz*Ll, Bsz*Ll, 3*Hh, 3*Hh, EMBP, 3*Hh, 0, 0, 0, 1, 0, 0);

    for (int t = 0; t < Ll; ++t) {
        gemm(hbf, WHHT, GH, bhh, Bsz, Bsz, 3*Hh, 3*Hh, Hh, 3*Hh, 0, 0, 0, 1, 0, 0);
        k_gru_gate<<<g1((long long)Bsz*Hh), blk, 0, stream>>>(GI, GH, hbuf, hbf, t);
    }

    // q projections from final hidden state (hbf = q_emb in bf16)
    gemm(hbf, WQT,  qrep, bq_net, Bsz, Bsz, Hh, Hh, Hh, Hh, 0, 0, 0, 1, 1, 0); // relu
    gemm(hbf, WSQT, sq,   nullptr,Bsz, Bsz, Hh, Hh, Hh, Hh, 0, 0, 0, 1, 2, 0); // tanh
    gemm(hbf, WVQT, vq,   nullptr,Bsz, Bsz, Hh, Hh, Hh, Hh, 0, 0, 0, 1, 2, 0); // tanh
    gemm(hbf, WAQT, aqb,  nullptr,Bsz, Bsz, Hh, Hh, Hh, Hh, 0, 0, 0, 1, 1, 0); // relu

    // ---- object path ----
    k_cvt<<<g1((long long)Bsz*Nn*OBJD), blk, 0, stream>>>(v, vbf, (long long)Bsz*Nn*OBJD);
    // v_proj = tanh(v @ wv_v): (25600 x 2048) * (2048 x 1024)
    gemm(vbf, WVVT, vproj, nullptr, Bsz*Nn, Bsz*Nn, Hh, Hh, OBJD, Hh, 0, 0, 0, 1, 2, 0);

    k_sproj<<<g1((long long)Bsz*128*Hh), blk, 0, stream>>>(bbox, ws_b, sq, Pbf, PQbf);
    k_vpack<<<g1((long long)Bsz*128*Hh), blk, 0, stream>>>(vproj, vq, VPbf, VPQbf);

    // rel logits = (PQ @ P^T) + (VPQ @ VP^T), batched over B
    gemm(PQbf,  Pbf,  rlog, nullptr, 128, Nn, 128, Nn, Hh, Nn,
         128LL*Hh, 128LL*Hh, (long long)Nn*Nn, Bsz, 0, 0);
    gemm(VPQbf, VPbf, rlog, nullptr, 128, Nn, 128, Nn, Hh, Nn,
         128LL*Hh, 128LL*Hh, (long long)Nn*Nn, Bsz, 0, 1);   // accumulate

    k_rel_softmax<<<g1((long long)Bsz*128*32), blk, 0, stream>>>(rlog, relbf);
    k_vT<<<g1((long long)Bsz*OBJD*128), blk, 0, stream>>>(v, vTbf);

    // v_nbr = rel @ v : batched (100 x 128pad) * (128pad x 2048)
    gemm(relbf, vTbf, vnbr, nullptr, 128, Nn, OBJD, OBJD, 128, OBJD,
         128LL*128, (long long)OBJD*128, (long long)Nn*OBJD, Bsz, 0, 0);

    // v2 = v + v_nbr  (bf16, reuse vbf)
    k_add_cvt<<<g1((long long)Bsz*Nn*OBJD), blk, 0, stream>>>(v, vnbr, vbf, (long long)Bsz*Nn*OBJD);

    // av = relu(v2 @ wa_v)  (reuse vproj as av)
    gemm(vbf, WAVT, vproj, nullptr, Bsz*Nn, Bsz*Nn, Hh, Hh, OBJD, Hh, 0, 0, 0, 1, 1, 0);

    k_att_logits<<<g1((long long)Bsz*Nn*32), blk, 0, stream>>>(vproj, aqb, wa_out, attl);
    k_att_softmax<<<g1((long long)Bsz*32), blk, 0, stream>>>(attl);
    k_vemb<<<g1((long long)Bsz*OBJD), blk, 0, stream>>>(v, vnbr, attl, vembbf);

    // v_repr = relu(v_emb @ wv_net + bv_net)
    gemm(vembbf, WVNT, vrep, bv_net, Bsz, Bsz, Hh, Hh, OBJD, Hh, 0, 0, 0, 1, 1, 0);

    // joint = q_repr * v_repr
    k_mul_cvt<<<g1((long long)Bsz*Hh), blk, 0, stream>>>(qrep, vrep, jntbf, (long long)Bsz*Hh);

    // hidden = relu(joint @ wc1 + bc1)
    gemm(jntbf, WC1T, hid, bc1, Bsz, Bsz, 2*Hh, 2*Hh, Hh, 2*Hh, 0, 0, 0, 1, 1, 0);
    k_cvt<<<g1((long long)Bsz*2*Hh), blk, 0, stream>>>(hid, hidbf, (long long)Bsz*2*Hh);

    // logits = hidden @ wc2 + bc2  -> d_out (256 x 3129)
    gemm(hidbf, WC2T, out, bc2, Bsz, Bsz, NANSP, NANSN, OBJD, NANSN, 0, 0, 0, 1, 0, 0);
}
